// GatherConvND_4724464026094
// MI455X (gfx1250) — matrix operand order repaired and checked by reading, not compiled, not run on hardware
//
#include <hip/hip_runtime.h>
#include <hip/hip_bf16.h>
#include <math.h>

// ---- problem constants (fixed by the reference) ----
#define BB     4
#define LL     2048
#define CC     1024
#define HH     16
#define KS     64      // max_kernel_size
#define SS     33      // grid samples
#define HALFS  16
#define MAXFREQ 16.0f
#define MINFREQ 1.0f
#define MAXREC  256.0f
#define MPOS   (BB*LL)           // 8192 rows

typedef __attribute__((ext_vector_type(16))) __bf16 v16bf;
typedef __attribute__((ext_vector_type(8)))  __bf16 v8bf;
typedef __attribute__((ext_vector_type(8)))  float  v8f;

__device__ __forceinline__ float silu_f(float v) { return v / (1.0f + __expf(-v)); }

__device__ __forceinline__ unsigned short f2bf(float f) {
    unsigned u = __float_as_uint(f);
    unsigned r = (u + 0x7FFFu + ((u >> 16) & 1u)) >> 16;   // RNE
    return (unsigned short)r;
}

// ================= Tensor Data Mover support =================
#if __has_builtin(__builtin_amdgcn_tensor_load_to_lds) && __has_builtin(__builtin_amdgcn_s_wait_tensorcnt)
#define USE_TDM 1
#endif

#ifdef USE_TDM
typedef __attribute__((ext_vector_type(4))) unsigned int u32x4;
typedef __attribute__((ext_vector_type(4))) int          i32x4;
typedef __attribute__((ext_vector_type(8))) int          i32x8;

__device__ __forceinline__ unsigned lds_byte_offset(void* p) {
    return (unsigned)(unsigned long long)(__attribute__((address_space(3))) void*)p;
}

// DMA a 64-row x 64-bf16 tile (128B rows) from row-major global [nRows x ldElems]
// into LDS with +16B padding per row (LDS row stride = 72 bf16 = 144B).
__device__ __forceinline__ void tdm_load_tile64x64(
    const unsigned short* g, int row0, int kb, int ldElems, int nRows, unsigned ldsOff) {
    unsigned long long ga =
        (unsigned long long)(size_t)(g + (size_t)row0 * ldElems + kb);
    u32x4 g0;
    g0.x = 1u;                                              // count=1, gather off
    g0.y = ldsOff;                                          // lds_addr (bytes)
    g0.z = (unsigned)ga;                                    // global_addr[31:0]
    g0.w = ((unsigned)(ga >> 32) & 0x01FFFFFFu) | 0x80000000u; // [56:32] | type=2
    i32x8 g1;
    // data_size=1(2B), pad_enable, pad_interval=4 (32 DWORDs=128B), pad_amount=3 (4 DWORDs=16B)
    g1[0] = (int)((1u << 16) | (1u << 20) | (4u << 22) | (3u << 25));
    g1[1] = (int)(((unsigned)ldElems & 0xFFFFu) << 16);     // tensor_dim0[15:0]
    g1[2] = (int)((((unsigned)ldElems >> 16) & 0xFFFFu) |
                  (((unsigned)nRows & 0xFFFFu) << 16));     // dim0[31:16] | dim1[15:0]
    g1[3] = (int)((((unsigned)nRows >> 16) & 0xFFFFu) |
                  (64u << 16));                             // dim1[31:16] | tile_dim0=64
    g1[4] = 64;                                             // tile_dim1=64, tile_dim2=0
    g1[5] = ldElems;                                        // tensor_dim0_stride[31:0]
    g1[6] = 0;
    g1[7] = 0;
    i32x4 z4 = {0, 0, 0, 0};
#if defined(__clang_major__) && __clang_major__ >= 23
    i32x8 z8 = {0, 0, 0, 0, 0, 0, 0, 0};
    __builtin_amdgcn_tensor_load_to_lds(g0, g1, z4, z4, z8, 0);
#else
    __builtin_amdgcn_tensor_load_to_lds(g0, g1, z4, z4, 0);
#endif
}
#endif // USE_TDM

// ---------------- fp32 -> bf16 conversion ----------------
__global__ void cvt_f32_bf16_kernel(const float* __restrict__ src,
                                    unsigned short* __restrict__ dst, int n) {
    int i = blockIdx.x * blockDim.x + threadIdx.x;
    if (i < n) dst[i] = f2bf(src[i]);
}

// ---------------- wave_proj: freq/phase averages ----------------
__global__ __launch_bounds__(128) void wave_proj_kernel(
    const float* __restrict__ x, const float* __restrict__ Wp,
    const float* __restrict__ bp,
    float* __restrict__ freq_avg, float* __restrict__ phase_avg) {
    int pos  = blockIdx.x * 4 + (threadIdx.x >> 5);   // wave index == position
    int lane = threadIdx.x & 31;
    const float4* xr = (const float4*)(x  + (size_t)pos  * CC);
    const float4* wr = (const float4*)(Wp + (size_t)lane * CC);
    float acc = 0.0f;
    for (int k = 0; k < CC / 4; ++k) {
        float4 a = xr[k], b = wr[k];
        acc += a.x * b.x + a.y * b.y + a.z * b.z + a.w * b.w;
    }
    float v = silu_f(acc + bp[lane]);
    float r;
    if (lane < HH) r = (1.0f / (1.0f + __expf(-v))) * (MAXFREQ - MINFREQ) + MINFREQ; // freq
    else           r = tanhf(v) * MAXFREQ;                                           // phase
    for (int off = 1; off < 16; off <<= 1) r += __shfl_xor(r, off, 32);
    if (lane == 0)  freq_avg[pos]  = r * (1.0f / HH);
    if (lane == 16) phase_avg[pos] = r * (1.0f / HH);
}

// ---------------- WMMA bf16 GEMM + SiLU (+optional bias) ----------------
// out[m,n] = silu( sum_k A[m,k]*W[n,k] + bias[n] ); A:[M,K], W:[N,K] bf16 row-major.
// Block tile 64x64 (4 waves x 16x64). TDM-staged double-buffered LDS tiles.
__global__ __launch_bounds__(128) void gemm_bf16_silu_kernel(
    const unsigned short* __restrict__ A, const unsigned short* __restrict__ W,
    const float* __restrict__ bias, float* __restrict__ out,
    int M, int N, int Kd) {
    __shared__ alignas(16) __bf16 As[2][64 * 72];
    __shared__ alignas(16) __bf16 Bs[2][64 * 72];

    const int tid  = threadIdx.x;
    const int lane = tid & 31;
    const int w    = tid >> 5;
    const int half = lane >> 4;
    const int r    = lane & 15;
    const int blockN = blockIdx.x * 64;
    const int blockM = blockIdx.y * 64;

    v8f acc[4] = {};
    const int niter = Kd >> 6;

#ifdef USE_TDM
    const bool issuer = (w == 0);   // wave 0 drives the Tensor Data Mover
    unsigned aOff[2], bOff[2];
    aOff[0] = lds_byte_offset(&As[0][0]);
    aOff[1] = lds_byte_offset(&As[1][0]);
    bOff[0] = lds_byte_offset(&Bs[0][0]);
    bOff[1] = lds_byte_offset(&Bs[1][0]);
    if (issuer) {
        tdm_load_tile64x64(A, blockM, 0, Kd, M, aOff[0]);
        tdm_load_tile64x64(W, blockN, 0, Kd, N, bOff[0]);
    }
#else
    const uint4* Ag = (const uint4*)A;
    const uint4* Bg = (const uint4*)W;
    const int rowU4 = Kd >> 3;
#endif

    for (int i = 0; i < niter; ++i) {
        const int cur = i & 1;
#ifdef USE_TDM
        if (issuer) {
            if (i + 1 < niter) {
                // prefetch next chunk into the other buffer, then wait for current
                tdm_load_tile64x64(A, blockM, (i + 1) << 6, Kd, M, aOff[cur ^ 1]);
                tdm_load_tile64x64(W, blockN, (i + 1) << 6, Kd, N, bOff[cur ^ 1]);
                __builtin_amdgcn_s_wait_tensorcnt(2);   // in-order: current tile done
            } else {
                __builtin_amdgcn_s_wait_tensorcnt(0);
            }
        }
        __syncthreads();
#else
        const int kb = i << 6;
        #pragma unroll
        for (int j = 0; j < 4; ++j) {
            int q  = tid + j * 128;
            int rw = q >> 3, cl = q & 7;
            uint4 va = Ag[(size_t)(blockM + rw) * rowU4 + (kb >> 3) + cl];
            *(uint4*)(&As[0][rw * 72 + cl * 8]) = va;
            uint4 vb = Bg[(size_t)(blockN + rw) * rowU4 + (kb >> 3) + cl];
            *(uint4*)(&Bs[0][rw * 72 + cl * 8]) = vb;
        }
        __syncthreads();
        const int curb = 0; (void)curb;
#endif
#ifdef USE_TDM
        const __bf16* Asb = &As[cur][0];
        const __bf16* Bsb = &Bs[cur][0];
#else
        const __bf16* Asb = &As[0][0];
        const __bf16* Bsb = &Bs[0][0];
#endif
        #pragma unroll
        for (int ks = 0; ks < 2; ++ks) {
            // A frag: lane<16 -> K 0..7 & 16..23 ; lane>=16 -> K 8..15 & 24..31
            const __bf16* arow = Asb + (w * 16 + r) * 72 + ks * 32;
            v8bf alo = *(const v8bf*)(arow + half * 8);
            v8bf ahi = *(const v8bf*)(arow + half * 8 + 16);
            v16bf afr = __builtin_shufflevector(alo, ahi,
                0,1,2,3,4,5,6,7,8,9,10,11,12,13,14,15);
            #pragma unroll
            for (int t = 0; t < 4; ++t) {
                // B frag: lane<16 -> K 0..7 & 16..23 ; lane>=16 -> K 8..15 & 24..31 (col = r)
                const __bf16* brow = Bsb + (t * 16 + r) * 72 + ks * 32;
                v8bf blo = *(const v8bf*)(brow + half * 8);
                v8bf bhi = *(const v8bf*)(brow + half * 8 + 16);
                v16bf bfr = __builtin_shufflevector(blo, bhi,
                    0,1,2,3,4,5,6,7,8,9,10,11,12,13,14,15);
                acc[t] = __builtin_amdgcn_wmma_f32_16x16x32_bf16(
                    false, afr, false, bfr, (short)0, acc[t], false, false);
            }
        }
        __syncthreads();   // tile fully consumed before it is overwritten
    }

    // epilogue: C/D layout -> VGPR v, lane<16: M=v, lane>=16: M=8+v, N=r
    #pragma unroll
    for (int t = 0; t < 4; ++t) {
        int n = blockN + t * 16 + r;
        float bv = bias ? bias[n] : 0.0f;
        int mb = blockM + w * 16 + half * 8;
        #pragma unroll
        for (int v = 0; v < 8; ++v) {
            float val = acc[t][v] + bv;
            out[(size_t)(mb + v) * N + n] = silu_f(val);
        }
    }
}

// ---------------- kernel interpolation + sample indices ----------------
__global__ __launch_bounds__(256) void interp_kernel(
    const float* __restrict__ km, const float* __restrict__ freq_avg,
    const float* __restrict__ phase_avg, float* __restrict__ kern,
    int* __restrict__ sidx) {
    int gid = blockIdx.x * blockDim.x + threadIdx.x;   // B*L*H threads
    int pos = gid >> 4;
    int h   = gid & 15;
    int l   = pos & (LL - 1);
    float freq = freq_avg[pos], phase = phase_avg[pos];
    const float* kmr = km + (size_t)pos * CC + h * KS;
    float kv[SS];
    float sum = 0.0f;
    #pragma unroll
    for (int s = 0; s < SS; ++s) {
        float rel = (float)(s - HALFS) * freq + phase;
        float sp  = (float)l + rel;
        bool valid = (sp >= 0.0f) && (sp < (float)LL);
        if (h == 0) {
            int idx = (int)sp;                       // trunc toward zero
            idx = idx < 0 ? 0 : (idx > LL - 1 ? LL - 1 : idx);
            sidx[pos * SS + s] = idx;
        }
        float norm = fminf(fmaxf((rel + MAXREC) / (2.0f * MAXREC), 0.0f), 1.0f);
        float fidx = norm * (float)(KS - 1);
        int ifl = (int)fidx; if (ifl > KS - 2) ifl = KS - 2;
        float wc = fidx - (float)ifl;
        float v = kmr[ifl] * (1.0f - wc) + kmr[ifl + 1] * wc;
        v = valid ? v : 0.0f;
        kv[s] = v;
        sum += v;
    }
    float inv = 1.0f / (sum + 1e-8f);
    float* kr = kern + ((size_t)pos * HH + h) * SS;
    #pragma unroll
    for (int s = 0; s < SS; ++s) kr[s] = kv[s] * inv;
}

// ---------------- gather + per-head conv, emit bf16 hidden ----------------
__global__ __launch_bounds__(256) void gatherconv_kernel(
    const float* __restrict__ x, const float* __restrict__ kern,
    const int* __restrict__ sidx, unsigned short* __restrict__ hidden) {
    int pos = blockIdx.x;            // b*L + l
    int tid = threadIdx.x;           // channel group: c = tid*4
    int b   = pos >> 11;             // / L
    int head = tid >> 4;             // (tid*4)/64
    const float* kr = kern + ((size_t)pos * HH + head) * SS;
    const int* sr = sidx + pos * SS;
    float4 acc = {0.f, 0.f, 0.f, 0.f};
    for (int s = 0; s < SS; ++s) {
        int idx = sr[s];
        if (s + 1 < SS) {   // hide the data-dependent gather latency
            __builtin_prefetch(x + (size_t)(b * LL + sr[s + 1]) * CC + tid * 4, 0, 0);
        }
        float wv = kr[s];
        const float4* xr = (const float4*)(x + (size_t)(b * LL + idx) * CC);
        float4 xv = xr[tid];
        acc.x += wv * xv.x; acc.y += wv * xv.y;
        acc.z += wv * xv.z; acc.w += wv * xv.w;
    }
    ushort4 o;
    o.x = f2bf(acc.x); o.y = f2bf(acc.y); o.z = f2bf(acc.z); o.w = f2bf(acc.w);
    *(ushort4*)(&hidden[(size_t)pos * CC + tid * 4]) = o;
}

// ---------------- workspace layout (bytes, 256B aligned) ----------------
static const size_t OFF_XB    = 0;                                        // x bf16  16 MB
static const size_t OFF_WK    = OFF_XB    + (size_t)MPOS * CC * 2;        // Wk bf16  2 MB
static const size_t OFF_WO    = OFF_WK    + (size_t)CC * CC * 2;          // Wo bf16  2 MB
static const size_t OFF_FREQ  = OFF_WO    + (size_t)CC * CC * 2;          // 32 KB
static const size_t OFF_PHASE = OFF_FREQ  + (size_t)MPOS * 4;             // 32 KB
static const size_t OFF_KM    = OFF_PHASE + (size_t)MPOS * 4;             // 32 MB
static const size_t OFF_KERN  = OFF_KM    + (size_t)MPOS * CC * 4;        // 16.5 MB
static const size_t OFF_SIDX  = OFF_KERN  + (size_t)MPOS * HH * SS * 4;   // 1 MB
static const size_t OFF_HID   = OFF_SIDX  + (size_t)MPOS * SS * 4;        // 16 MB

extern "C" void kernel_launch(void* const* d_in, const int* in_sizes, int n_in,
                              void* d_out, int out_size, void* d_ws, size_t ws_size,
                              hipStream_t stream) {
    const float* x  = (const float*)d_in[0];
    const float* Wp = (const float*)d_in[1];
    const float* bp = (const float*)d_in[2];
    const float* Wk = (const float*)d_in[3];
    const float* bk = (const float*)d_in[4];
    const float* Wo = (const float*)d_in[5];
    float* out = (float*)d_out;

    char* ws = (char*)d_ws;
    unsigned short* xb   = (unsigned short*)(ws + OFF_XB);
    unsigned short* wkb  = (unsigned short*)(ws + OFF_WK);
    unsigned short* wob  = (unsigned short*)(ws + OFF_WO);
    float* freq_avg      = (float*)(ws + OFF_FREQ);
    float* phase_avg     = (float*)(ws + OFF_PHASE);
    float* km            = (float*)(ws + OFF_KM);
    float* kern          = (float*)(ws + OFF_KERN);
    int*   sidx          = (int*)(ws + OFF_SIDX);
    unsigned short* hid  = (unsigned short*)(ws + OFF_HID);

    // 1) fp32 -> bf16 conversions (x, Wk, Wo)
    cvt_f32_bf16_kernel<<<(MPOS * CC) / 256, 256, 0, stream>>>(x, xb, MPOS * CC);
    cvt_f32_bf16_kernel<<<(CC * CC) / 256, 256, 0, stream>>>(Wk, wkb, CC * CC);
    cvt_f32_bf16_kernel<<<(CC * CC) / 256, 256, 0, stream>>>(Wo, wob, CC * CC);

    // 2) wave projection -> freq/phase averages (one wave per position)
    wave_proj_kernel<<<MPOS / 4, 128, 0, stream>>>(x, Wp, bp, freq_avg, phase_avg);

    // 3) kernel_proj GEMM: km = silu(x @ Wk^T + bk)   [8192 x 1024 x 1024]
    {
        dim3 grid(CC / 64, MPOS / 64);
        gemm_bf16_silu_kernel<<<grid, 128, 0, stream>>>(xb, wkb, bk, km, MPOS, CC, CC);
    }

    // 4) kernel interpolation + sample indices
    interp_kernel<<<(MPOS * HH) / 256, 256, 0, stream>>>(km, freq_avg, phase_avg, kern, sidx);

    // 5) gather + per-head conv -> bf16 hidden
    gatherconv_kernel<<<MPOS, 256, 0, stream>>>(x, kern, sidx, hid);

    // 6) out_proj GEMM: out = silu(hidden @ Wo^T)   [8192 x 1024 x 1024]
    {
        dim3 grid(CC / 64, MPOS / 64);
        gemm_bf16_silu_kernel<<<grid, 128, 0, stream>>>(hid, wob, nullptr, out, MPOS, CC, CC);
    }
}
